// CausalPredictiveMemoryMachine_36051955483063
// MI455X (gfx1250) — matrix operand order repaired and checked by reading, not compiled, not run on hardware
//
#include <hip/hip_runtime.h>
#include <math.h>

typedef __attribute__((ext_vector_type(16))) _Float16 v16h;
typedef __attribute__((ext_vector_type(8)))  float    v8f;
typedef __attribute__((ext_vector_type(4)))  float    v4f;

#define NUM_SYMBOLS 64
#define NUM_VALUES  32
#define D_MODEL     64
#define TPW         4     // tiles (of 16 tokens) per wave in kernel 1

__device__ __forceinline__ float gelu_exact(float x) {
    return 0.5f * x * (1.0f + erff(x * 0.70710678118654752440f));
}

// ---------------------------------------------------------------------------
// Kernel 1: per-token gate MLPs via WMMA f16 (the bandwidth/FLOP-heavy phase).
// Each wave processes TPW consecutive 16-token tiles. All W1
// B-fragments (16 x v16h = 128 VGPRs), biases, and W2 columns are hoisted
// into registers once per wave, so the steady-state tile loop is pure
// global B128 loads + f32->f16 cvt + 16 WMMAs + epilogue (no LDS traffic).
// ---------------------------------------------------------------------------
__global__ __launch_bounds__(256) void cpm_gates_wmma(
    const float* __restrict__ evidence,      // [BT, 64] fp32
    const float* __restrict__ mgW1, const float* __restrict__ mgb1,
    const float* __restrict__ mgW2, const float* __restrict__ mgb2,
    const float* __restrict__ sgW1, const float* __restrict__ sgb1,
    const float* __restrict__ sgW2, const float* __restrict__ sgb2,
    float* __restrict__ map_g, float* __restrict__ step_g,
    int ntiles)
{
    // W1 staged transposed ([out][in]) in f16 so each lane's B-fragment is
    // one contiguous 32-byte LDS read (done once per wave, then registers).
    __shared__ __align__(32) _Float16 sW[2][D_MODEL * D_MODEL];
    __shared__ float sB1[2][D_MODEL];
    __shared__ float sW2v[2][D_MODEL];

    for (int i = threadIdx.x; i < D_MODEL * D_MODEL; i += 256) {
        int n = i >> 6, k = i & 63;              // n = out, k = in
        sW[0][n * 64 + k] = (_Float16)mgW1[k * 64 + n];
        sW[1][n * 64 + k] = (_Float16)sgW1[k * 64 + n];
    }
    if (threadIdx.x < D_MODEL) {
        sB1[0][threadIdx.x]  = mgb1[threadIdx.x];
        sB1[1][threadIdx.x]  = sgb1[threadIdx.x];
        sW2v[0][threadIdx.x] = mgW2[threadIdx.x];
        sW2v[1][threadIdx.x] = sgW2[threadIdx.x];
    }
    __syncthreads();

    const int wave = threadIdx.x >> 5;
    const int lane = threadIdx.x & 31;
    const int m    = lane & 15;          // row within tile / N within N-tile
    const int h    = lane >> 4;          // lane-group (K striping)

    // --- hoist B-fragments + epilogue scalars into registers -------------
    // B (32x16 f16) layout: lane group h holds K = 16h + e, N = m.
    v16h bf[2][4][2];
    float bias_r[2][4], w2_r[2][4];
#pragma unroll
    for (int mlp = 0; mlp < 2; ++mlp) {
#pragma unroll
        for (int nt = 0; nt < 4; ++nt) {
            bf[mlp][nt][0] = *(const v16h*)&sW[mlp][(16 * nt + m) * 64 + 16 * h];
            bf[mlp][nt][1] = *(const v16h*)&sW[mlp][(16 * nt + m) * 64 + 32 + 16 * h];
            bias_r[mlp][nt] = sB1[mlp][16 * nt + m];
            w2_r[mlp][nt]   = sW2v[mlp][16 * nt + m];
        }
    }
    const float b2_r[2] = {mgb2[0], sgb2[0]};

    const int tile0 = (blockIdx.x * 8 + wave) * TPW;

    for (int it = 0; it < TPW; ++it) {
        const int tile = tile0 + it;
        if (tile >= ntiles) break;

        const long base = (long)tile * 16;
        const float* __restrict__ row = evidence + (base + m) * (long)D_MODEL;

        // Warm L2 for the next tile this wave will touch.
        __builtin_prefetch(row + 16L * D_MODEL, 0, 1);

        // --- A fragments (16x32 f16): lane (m,h) holds
        // row m, K = 16h + e in elems 0..15 (matches B).
        const v4f* r4 = (const v4f*)row;
        v4f q0 = r4[4 * h + 0],  q1 = r4[4 * h + 1];    // K 16h..        (kt=0)
        v4f q2 = r4[4 * h + 2],  q3 = r4[4 * h + 3];    // K 16h+8..      (kt=0)
        v4f q4 = r4[4 * h + 8],  q5 = r4[4 * h + 9];    // K 32+16h..     (kt=1)
        v4f q6 = r4[4 * h + 10], q7 = r4[4 * h + 11];   // K 40+16h..     (kt=1)

        v16h a0, a1;
#pragma unroll
        for (int e = 0; e < 4; ++e) {
            a0[e]      = (_Float16)q0[e];
            a0[e + 4]  = (_Float16)q1[e];
            a0[e + 8]  = (_Float16)q2[e];
            a0[e + 12] = (_Float16)q3[e];
            a1[e]      = (_Float16)q4[e];
            a1[e + 4]  = (_Float16)q5[e];
            a1[e + 8]  = (_Float16)q6[e];
            a1[e + 12] = (_Float16)q7[e];
        }

#pragma unroll
        for (int mlp = 0; mlp < 2; ++mlp) {
            v8f acc[4] = {v8f{}, v8f{}, v8f{}, v8f{}};
#pragma unroll
            for (int nt = 0; nt < 4; ++nt) {
                acc[nt] = __builtin_amdgcn_wmma_f32_16x16x32_f16(
                    false, a0, false, bf[mlp][nt][0], (short)0, acc[nt], false, false);
                acc[nt] = __builtin_amdgcn_wmma_f32_16x16x32_f16(
                    false, a1, false, bf[mlp][nt][1], (short)0, acc[nt], false, false);
            }

            // Epilogue: bias + exact GELU + dot with W2 column, per lane's N.
            float p[8] = {0.f, 0.f, 0.f, 0.f, 0.f, 0.f, 0.f, 0.f};
#pragma unroll
            for (int nt = 0; nt < 4; ++nt) {
                const float bi = bias_r[mlp][nt];
                const float w2 = w2_r[mlp][nt];
#pragma unroll
                for (int r = 0; r < 8; ++r) {
                    float x = acc[nt][r] + bi;
                    p[r] += gelu_exact(x) * w2;
                }
            }

            // Reduce over the 16-lane N group (C layout: lanes 0-15 hold
            // rows 0..7 in VGPRs 0..7; lanes 16-31 hold rows 8..15).
#pragma unroll
            for (int r = 0; r < 8; ++r) {
                p[r] += __shfl_xor(p[r], 1, 16);
                p[r] += __shfl_xor(p[r], 2, 16);
                p[r] += __shfl_xor(p[r], 4, 16);
                p[r] += __shfl_xor(p[r], 8, 16);
            }

            if (m < 8) {
                const float dot = p[m] + b2_r[mlp];
                const float g   = 1.0f / (1.0f + expf(-dot));
                const long  tok = base + 8 * h + m;
                if (mlp == 0) map_g[tok]  = g;
                else          step_g[tok] = g;
            }
        }
    }
}

// ---------------------------------------------------------------------------
// Kernel 2: per-batch memories in LDS (scatter) + 3-step graph walk + heads.
// One 256-thread block per batch; all memories stay in ~25 KB of LDS.
// ---------------------------------------------------------------------------
__global__ __launch_bounds__(256) void cpm_walk_heads(
    const int* __restrict__ em,
    const int* __restrict__ src_i,  const int* __restrict__ src_v,
    const int* __restrict__ tsym_i, const int* __restrict__ tsym_v,
    const int* __restrict__ tval_i, const int* __restrict__ tval_v,
    const int* __restrict__ q_i,    const int* __restrict__ q_v,
    const float* __restrict__ symbol_emb,   // [64,64]
    const float* __restrict__ value_emb,    // [32,64]
    const float* __restrict__ sfW1, const float* __restrict__ sfb1,
    const float* __restrict__ sfW2, const float* __restrict__ sfb2,
    const float* __restrict__ ohW1, const float* __restrict__ ohb1,
    const float* __restrict__ ohW2, const float* __restrict__ ohb2,
    const float* __restrict__ map_g, const float* __restrict__ step_g,
    float* __restrict__ out, int B, int T)
{
    const int b   = blockIdx.x;
    const int tid = threadIdx.x;

    __shared__ float s_map[NUM_SYMBOLS * 33];   // padded stride 33
    __shared__ float s_step[NUM_SYMBOLS * 65];  // padded stride 65
    __shared__ float s_walk[NUM_SYMBOLS];
    __shared__ float s_new[NUM_SYMBOLS];
    __shared__ float s_accv[NUM_VALUES];
    __shared__ float s_accs[D_MODEL];
    __shared__ float s_gs[2 * D_MODEL];
    __shared__ float s_h[2 * D_MODEL];

    for (int i = tid; i < NUM_SYMBOLS * 33; i += 256) s_map[i] = 0.f;
    for (int i = tid; i < NUM_SYMBOLS * 65; i += 256) s_step[i] = 0.f;
    __syncthreads();

    // --- scatter masked gains (order-independent additive updates) ---
    if (tid < T) {
        const long g  = (long)b * T + tid;
        const int  e  = em[g];
        const bool sv = src_v[g] != 0;
        const int  s  = min(max(src_i[g], 0), NUM_SYMBOLS - 1);
        if (((e == 1) || (e == 2)) && sv && (tval_v[g] != 0)) {
            const int v = min(max(tval_i[g], 0), NUM_VALUES - 1);
            atomicAdd(&s_map[s * 33 + v], map_g[g]);
        }
        if ((e == 3) && sv && (tsym_v[g] != 0)) {
            const int t2 = min(max(tsym_i[g], 0), NUM_SYMBOLS - 1);
            atomicAdd(&s_step[s * 65 + t2], step_g[g]);
        }
    }
    if (tid < NUM_SYMBOLS) {
        const int q = min(max(q_i[b], 0), NUM_SYMBOLS - 1);
        s_walk[tid] = (tid == q) ? (float)q_v[b] : 0.f;
    }
    __syncthreads();

    // --- initial accumulators ---
    if (tid < NUM_VALUES) {
        float a = 0.f;
        for (int s = 0; s < NUM_SYMBOLS; ++s) a += s_walk[s] * s_map[s * 33 + tid];
        s_accv[tid] = a;
    } else if (tid >= 64 && tid < 128) {
        const int d = tid - 64;
        float a = 0.f;
        for (int s = 0; s < NUM_SYMBOLS; ++s) a += s_walk[s] * symbol_emb[s * 64 + d];
        s_accs[d] = a;
    }
    __syncthreads();

    // --- 3-step walk ---
    for (int step = 0; step < 3; ++step) {
        if (tid < NUM_SYMBOLS) {
            float a = 0.f;
            for (int s = 0; s < NUM_SYMBOLS; ++s) a += s_walk[s] * s_step[s * 65 + tid];
            s_new[tid] = a;
        }
        __syncthreads();
        if (tid < NUM_SYMBOLS) s_walk[tid] = s_new[tid];
        __syncthreads();
        if (tid < NUM_VALUES) {
            float a = s_accv[tid];
            for (int s = 0; s < NUM_SYMBOLS; ++s) a += s_walk[s] * s_map[s * 33 + tid];
            s_accv[tid] = a;
        } else if (tid >= 64 && tid < 128) {
            const int d = tid - 64;
            float a = s_accs[d];
            for (int s = 0; s < NUM_SYMBOLS; ++s) a += s_walk[s] * symbol_emb[s * 64 + d];
            s_accs[d] = a;
        }
        __syncthreads();
    }

    // --- graph state = [acc_symbols | acc_values @ value_emb] ---
    if (tid < D_MODEL) {
        s_gs[tid] = s_accs[tid];
    } else if (tid < 128) {
        const int d = tid - 64;
        float a = 0.f;
        for (int v = 0; v < NUM_VALUES; ++v) a += s_accv[v] * value_emb[v * 64 + d];
        s_gs[D_MODEL + d] = a;
    }
    __syncthreads();

    // --- two head MLPs in parallel (hidden layer) ---
    if (tid < D_MODEL) {                      // output_head hidden
        float a = ohb1[tid];
        for (int i = 0; i < 2 * D_MODEL; ++i) a += s_gs[i] * ohW1[i * 64 + tid];
        s_h[tid] = gelu_exact(a);
    } else if (tid < 128) {                   // slot_feedback hidden
        const int j = tid - 64;
        float a = sfb1[j];
        for (int i = 0; i < 2 * D_MODEL; ++i) a += s_gs[i] * sfW1[i * 64 + j];
        s_h[D_MODEL + j] = gelu_exact(a);
    }
    __syncthreads();

    // --- head outputs ---
    if (tid < NUM_VALUES) {                   // logits [B,32] first in d_out
        float a = ohb2[tid];
        for (int j = 0; j < D_MODEL; ++j) a += s_h[j] * ohW2[j * NUM_VALUES + tid];
        out[(long)b * NUM_VALUES + tid] = a;
    } else if (tid >= 64 && tid < 128) {      // feedback [B,64] second
        const int d = tid - 64;
        float a = sfb2[d];
        for (int j = 0; j < D_MODEL; ++j) a += s_h[D_MODEL + j] * sfW2[j * D_MODEL + d];
        out[(long)B * NUM_VALUES + (long)b * D_MODEL + d] = a;
    }
}

// ---------------------------------------------------------------------------
extern "C" void kernel_launch(void* const* d_in, const int* in_sizes, int n_in,
                              void* d_out, int out_size, void* d_ws, size_t ws_size,
                              hipStream_t stream)
{
    const float* evidence = (const float*)d_in[0];
    const int*   em       = (const int*)d_in[1];
    const int*   src_i    = (const int*)d_in[2];
    const int*   src_v    = (const int*)d_in[3];
    const int*   tsym_i   = (const int*)d_in[4];
    const int*   tsym_v   = (const int*)d_in[5];
    const int*   tval_i   = (const int*)d_in[6];
    const int*   tval_v   = (const int*)d_in[7];
    const int*   q_i      = (const int*)d_in[8];
    const int*   q_v      = (const int*)d_in[9];
    const float* symbol_emb = (const float*)d_in[10];
    const float* value_emb  = (const float*)d_in[11];
    const float* mgW1 = (const float*)d_in[12];
    const float* mgb1 = (const float*)d_in[13];
    const float* mgW2 = (const float*)d_in[14];
    const float* mgb2 = (const float*)d_in[15];
    const float* sgW1 = (const float*)d_in[16];
    const float* sgb1 = (const float*)d_in[17];
    const float* sgW2 = (const float*)d_in[18];
    const float* sgb2 = (const float*)d_in[19];
    const float* sfW1 = (const float*)d_in[20];
    const float* sfb1 = (const float*)d_in[21];
    const float* sfW2 = (const float*)d_in[22];
    const float* sfb2 = (const float*)d_in[23];
    const float* ohW1 = (const float*)d_in[24];
    const float* ohb1 = (const float*)d_in[25];
    const float* ohW2 = (const float*)d_in[26];
    const float* ohb2 = (const float*)d_in[27];

    const int  B  = in_sizes[8];              // query_idx is [B]
    const int  T  = in_sizes[1] / B;          // event_marker is [B,T]
    const long BT = (long)B * T;

    float* map_g  = (float*)d_ws;             // [BT]
    float* step_g = map_g + BT;               // [BT]

    const int ntiles  = (int)(BT / 16);
    const int blocksA = (ntiles + 8 * TPW - 1) / (8 * TPW);

    cpm_gates_wmma<<<blocksA, 256, 0, stream>>>(
        evidence, mgW1, mgb1, mgW2, mgb2, sgW1, sgb1, sgW2, sgb2,
        map_g, step_g, ntiles);

    cpm_walk_heads<<<B, 256, 0, stream>>>(
        em, src_i, src_v, tsym_i, tsym_v, tval_i, tval_v, q_i, q_v,
        symbol_emb, value_emb, sfW1, sfb1, sfW2, sfb2,
        ohW1, ohb1, ohW2, ohb2, map_g, step_g,
        (float*)d_out, B, T);
}
